// GNN_LinearAttn_2070174236743
// MI455X (gfx1250) — matrix operand order repaired and checked by reading, not compiled, not run on hardware
//
#include <hip/hip_runtime.h>

typedef __attribute__((ext_vector_type(16))) __bf16        v16bf;
typedef __attribute__((ext_vector_type(8)))  float         v8f;
typedef __attribute__((ext_vector_type(4)))  unsigned int  u32x4;
typedef __attribute__((ext_vector_type(4)))  float         f32x4;
typedef __attribute__((ext_vector_type(8)))  int           i32x8;
typedef __attribute__((ext_vector_type(4)))  int           i32x4;

#define BB 8
#define NN 2048
#define DD 128
#define OO 128

__device__ __forceinline__ float sigmoidf_(float x) { return 1.0f / (1.0f + __expf(-x)); }

// ---------------------------------------------------------------------------
// WMMA fragment loaders (wave32, 16x16x32 bf16).
// A-matrix 16x32: lane m (m = lane&15) holds row m; lanes<16: K = {0..7,16..23},
// lanes>=16: K = {8..15,24..31}  (ISA 7.12.2 "16-bit A-Matrix 16x32").
__device__ __forceinline__ v16bf load_afrag(const __bf16* src, int ld) {
  int lane = threadIdx.x & 31;
  int row  = lane & 15;
  int hi   = lane >> 4;
  const __bf16* p = src + row * ld + hi * 8;
  union { v16bf v; u32x4 q[2]; } u;
  u.q[0] = *(const u32x4*)(p);
  u.q[1] = *(const u32x4*)(p + 16);
  return u.v;
}

// B-matrix 32x16: lane n holds column n; lanes<16: K={0..7,16..23}, lanes>=16: K={8..15,24..31}.
// Source is row-major [col][k] (the transposed operand), ld elements.
__device__ __forceinline__ v16bf load_bfrag(const __bf16* srcT, int ld) {
  int lane = threadIdx.x & 31;
  int col  = lane & 15;
  int hi   = lane >> 4;
  const __bf16* p = srcT + col * ld + hi * 8;
  union { v16bf v; u32x4 q[2]; } u;
  u.q[0] = *(const u32x4*)(p);
  u.q[1] = *(const u32x4*)(p + 16);
  return u.v;
}

__device__ __forceinline__ v8f wmma_bf16(v16bf a, v16bf b, v8f c) {
  return __builtin_amdgcn_wmma_f32_16x16x32_bf16(false, a, false, b, (short)0, c, false, false);
}

// ---------------------------------------------------------------------------
// TDM: issue a 2-D tensor tile load (global -> LDS).  D# per ISA ch.8:
// group0 = {flags(count=1), lds_addr, global_addr lo, global_addr hi | type=2}
// group1 packs data_size=2B, tensor_dim0/1, tile_dim0/1, tensor_dim0_stride.
// Tracked on TENSORcnt; in-order per wave.  (6-arg builtin on this toolchain.)
__device__ __forceinline__ void tdm_load_2d(const void* gaddr, unsigned lds_off,
                                            unsigned tile_d0, unsigned tile_d1,
                                            unsigned tensor_d0, unsigned tensor_d1,
                                            unsigned stride0) {
  unsigned long long ga = (unsigned long long)gaddr;
  u32x4 g0;
  g0.x = 1u;                                        // count=1 (valid), user mode
  g0.y = lds_off;                                   // LDS byte address
  g0.z = (unsigned)(ga & 0xFFFFFFFFu);              // global_addr[31:0]
  g0.w = (unsigned)((ga >> 32) & 0x01FFFFFFu) | (2u << 30);  // addr[56:32] | type=2
  i32x8 g1;
  g1[0] = (int)(1u << 16);                          // data_size code 1 = 2 bytes
  g1[1] = (int)((tensor_d0 & 0xFFFFu) << 16);       // tensor_dim0[15:0]
  g1[2] = (int)((tensor_d0 >> 16) | ((tensor_d1 & 0xFFFFu) << 16));
  g1[3] = (int)((tensor_d1 >> 16) | (tile_d0 << 16));
  g1[4] = (int)tile_d1;                             // tile_dim1 (tile_dim2=0)
  g1[5] = (int)stride0;                             // tensor_dim0_stride[31:0]
  g1[6] = 0;
  g1[7] = 0;
  i32x4 z4 = {0, 0, 0, 0};
  i32x8 z8 = {0, 0, 0, 0, 0, 0, 0, 0};
  __builtin_amdgcn_tensor_load_to_lds(g0, g1, z4, z4, z8, 0);
}

// ---------------------------------------------------------------------------
// Prep: transpose + bf16-convert the three 128x128 weight matrices.
__global__ __launch_bounds__(256) void k_prep(const float* __restrict__ Wqk,
                                              const float* __restrict__ Wl,
                                              const float* __restrict__ Wr,
                                              __bf16* __restrict__ WqkT,
                                              __bf16* __restrict__ WlT,
                                              __bf16* __restrict__ WrT) {
  const float* src = (blockIdx.x == 0) ? Wqk : (blockIdx.x == 1) ? Wl : Wr;
  __bf16* dst      = (blockIdx.x == 0) ? WqkT : (blockIdx.x == 1) ? WlT : WrT;
  for (int i = threadIdx.x; i < DD * DD; i += 256) {
    int n = i >> 7, k = i & (DD - 1);
    dst[i] = (__bf16)src[k * DD + n];
  }
}

// ---------------------------------------------------------------------------
// Degree: deg[b,n] = sum_m A[b,n,m].  One wave per row.
__global__ __launch_bounds__(256) void k_deg(const float* __restrict__ A,
                                             float* __restrict__ deg) {
  int wave = threadIdx.x >> 5, lane = threadIdx.x & 31;
  size_t row = (size_t)blockIdx.x * 8 + wave;
  const f32x4* p = (const f32x4*)(A + row * NN);
  float s = 0.f;
#pragma unroll 4
  for (int i = lane; i < NN / 4; i += 32) { f32x4 v = p[i]; s += v.x + v.y + v.z + v.w; }
#pragma unroll
  for (int m = 16; m > 0; m >>= 1) s += __shfl_xor(s, m, 32);
  if (lane == 0) deg[row] = s;
}

// ---------------------------------------------------------------------------
// xg = x * sigmoid(deg * W_d + b_d), bf16 row-major + transposed per batch.
__global__ __launch_bounds__(256) void k_xg(const float* __restrict__ x,
                                            const float* __restrict__ deg,
                                            const float* __restrict__ Wd,
                                            const float* __restrict__ bd,
                                            __bf16* __restrict__ xg,
                                            __bf16* __restrict__ xgT) {
  size_t i = ((size_t)blockIdx.x * 256 + threadIdx.x) * 4;
  int d = (int)(i & (DD - 1));
  size_t row = i >> 7;
  int n = (int)(row & (NN - 1));
  int b = (int)(row >> 11);
  float dg = deg[row];
  f32x4 xv = *(const f32x4*)(x + i);
  f32x4 wd = *(const f32x4*)(Wd + d);
  f32x4 bv = *(const f32x4*)(bd + d);
  union { __bf16 h[4]; unsigned long long u; } o;
  o.h[0] = (__bf16)(xv.x * sigmoidf_(dg * wd.x + bv.x));
  o.h[1] = (__bf16)(xv.y * sigmoidf_(dg * wd.y + bv.y));
  o.h[2] = (__bf16)(xv.z * sigmoidf_(dg * wd.z + bv.z));
  o.h[3] = (__bf16)(xv.w * sigmoidf_(dg * wd.w + bv.w));
  *(unsigned long long*)(xg + i) = o.u;
  __bf16* tb = xgT + ((size_t)b * DD + d) * NN + n;
  tb[0] = o.h[0]; tb[NN] = o.h[1]; tb[2 * NN] = o.h[2]; tb[3 * NN] = o.h[3];
}

// ---------------------------------------------------------------------------
// QK = sigmoid(xg @ W_qk + b_qk) in bf16.  128 rows per block, 16 per wave.
__global__ __launch_bounds__(256) void k_qk(const __bf16* __restrict__ xg,
                                            const __bf16* __restrict__ WqkT,
                                            const float* __restrict__ bqk,
                                            __bf16* __restrict__ QK) {
  __shared__ __bf16 sW[DD * DD];
  for (int i = threadIdx.x; i < DD * DD / 8; i += 256)
    ((u32x4*)sW)[i] = ((const u32x4*)WqkT)[i];
  __syncthreads();
  int wave = threadIdx.x >> 5, lane = threadIdx.x & 31;
  int ccol = lane & 15, crow = (lane >> 4) * 8;
  size_t m0 = (size_t)blockIdx.x * 128 + wave * 16;
  v16bf af[4];
#pragma unroll
  for (int kk = 0; kk < 4; kk++) af[kk] = load_afrag(xg + m0 * DD + kk * 32, DD);
#pragma unroll
  for (int dt = 0; dt < 8; dt++) {
    v8f c = {0.f, 0.f, 0.f, 0.f, 0.f, 0.f, 0.f, 0.f};
#pragma unroll
    for (int kk = 0; kk < 4; kk++)
      c = wmma_bf16(af[kk], load_bfrag(sW + (dt * 16) * DD + kk * 32, DD), c);
    float bias = bqk[dt * 16 + ccol];
    __bf16* op = QK + (m0 + crow) * DD + dt * 16 + ccol;
#pragma unroll
    for (int r = 0; r < 8; r++)
      op[(size_t)r * DD] = (__bf16)sigmoidf_(c[r] + bias);
  }
}

// ---------------------------------------------------------------------------
// Fused masked attention + aggregation, flash style, TDM-staged operands.
// Grid: (b, row-tile, half).  Each block covers 128 rows x 1024 neighbors,
// emitting unnormalized f32 partial agg + partial rowsum.
// Double-buffered LDS stages of QK (64x128) and xgT (128x64) via TDM.
#define MT      64                  // neighbor columns per stage
#define NSTAGE  (NN / 2 / MT)       // 16 stages per half
#define OFF_QK(buf) ((buf) * 32768u)
#define OFF_XG(buf) ((buf) * 32768u + 16384u)
#define OFF_SCR(w)  (65536u + (w) * 1024u)

__global__ __launch_bounds__(256) void k_attn(const __bf16* __restrict__ QK,
                                              const __bf16* __restrict__ xgT,
                                              const float* __restrict__ A,
                                              float* __restrict__ aggpart,
                                              float* __restrict__ rspart) {
  constexpr float SCALE = 0.08838834764831845f;  // 1/sqrt(128)
  __shared__ __attribute__((aligned(16))) unsigned char smem[73728];
  int wave = threadIdx.x >> 5, lane = threadIdx.x & 31;
  int ccol = lane & 15, crow = (lane >> 4) * 8;
  int half = blockIdx.x & 1;
  int rt   = (blockIdx.x >> 1) & 15;
  int b    = blockIdx.x >> 5;
  int m0   = rt * 128 + wave * 16;
  int mt0  = half * (NN / 2);
  const __bf16* QKb  = QK  + (size_t)b * NN * DD;
  const __bf16* xgTb = xgT + (size_t)b * DD * NN;
  const float*  Ab   = A   + (size_t)b * NN * NN;
  __bf16* scr = (__bf16*)(smem + OFF_SCR(wave));

  v16bf qf[4];
#pragma unroll
  for (int kk = 0; kk < 4; kk++) qf[kk] = load_afrag(QKb + (size_t)m0 * DD + kk * 32, DD);

  v8f acc[8];
  v8f rs = {0.f, 0.f, 0.f, 0.f, 0.f, 0.f, 0.f, 0.f};
#pragma unroll
  for (int dt = 0; dt < 8; dt++) acc[dt] = rs;

  // prologue: stage 0 -> buffer 0 (wave 0 drives the TDM)
  if (wave == 0) {
    tdm_load_2d(QKb + (size_t)mt0 * DD, OFF_QK(0), DD, MT, DD, NN, DD);
    tdm_load_2d(xgTb + mt0,             OFF_XG(0), MT, DD, NN, DD, NN);
  }

  for (int it = 0; it < NSTAGE; it++) {
    int cur = it & 1;
    int mt  = mt0 + it * MT;
    __syncthreads();  // everyone done computing on the buffer we are about to refill
    if (wave == 0) {
      if (it + 1 < NSTAGE) {
        int mtn = mt + MT;
        tdm_load_2d(QKb + (size_t)mtn * DD, OFF_QK(cur ^ 1), DD, MT, DD, NN, DD);
        tdm_load_2d(xgTb + mtn,             OFF_XG(cur ^ 1), MT, DD, NN, DD, NN);
        __builtin_amdgcn_s_wait_tensorcnt(2);  // stage `it` complete, next in flight
      } else {
        __builtin_amdgcn_s_wait_tensorcnt(0);
      }
    }
    __syncthreads();  // stage `it` visible to all waves
    const __bf16* sQK = (const __bf16*)(smem + OFF_QK(cur));
    const __bf16* sXG = (const __bf16*)(smem + OFF_XG(cur));

#pragma unroll
    for (int h = 0; h < 2; h++) {          // two 32-col halves of the stage
#pragma unroll
      for (int j = 0; j < 2; j++) {        // 16-col score subtiles
        v8f S = {0.f, 0.f, 0.f, 0.f, 0.f, 0.f, 0.f, 0.f};
#pragma unroll
        for (int kk = 0; kk < 4; kk++)
          S = wmma_bf16(qf[kk], load_bfrag(sQK + (h * 32 + j * 16) * DD + kk * 32, DD), S);
        const float* Arow = Ab + (size_t)(m0 + crow) * NN + mt + h * 32 + j * 16 + ccol;
        __builtin_prefetch((const void*)(Arow + MT), 0, 0);  // next stage's mask
#pragma unroll
        for (int r = 0; r < 8; r++) {
          float s = S[r] * SCALE * Arow[(size_t)r * NN];     // adjacency mask
          rs[r] += s;
          scr[(crow + r) * 32 + j * 16 + ccol] = (__bf16)s;
        }
      }
      asm volatile("s_wait_dscnt 0" ::: "memory");  // per-wave scratch stores -> loads
      v16bf sfrag = load_afrag(scr, 32);
#pragma unroll
      for (int dt = 0; dt < 8; dt++)
        acc[dt] = wmma_bf16(sfrag, load_bfrag(sXG + (dt * 16) * MT + h * 32, MT), acc[dt]);
    }
  }

  // partial rowsum: reduce across the 16-lane column group
#pragma unroll
  for (int r = 0; r < 8; r++) {
    float v = rs[r];
    v += __shfl_xor(v, 1, 32);
    v += __shfl_xor(v, 2, 32);
    v += __shfl_xor(v, 4, 32);
    v += __shfl_xor(v, 8, 32);
    rs[r] = v;
  }
  size_t rowbase = (size_t)b * NN + m0;
#pragma unroll
  for (int r = 0; r < 8; r++)
    if (ccol == 0) rspart[(rowbase + crow + r) * 2 + half] = rs[r];
#pragma unroll
  for (int dt = 0; dt < 8; dt++)
#pragma unroll
    for (int r = 0; r < 8; r++)
      aggpart[((rowbase + crow + r) * 2 + half) * DD + dt * 16 + ccol] = acc[dt][r];
}

// ---------------------------------------------------------------------------
// Combine the two neighbor-halves: agg = (p0+p1)/(rs0+rs1+1e-6) -> bf16.
__global__ __launch_bounds__(256) void k_comb(const float* __restrict__ aggpart,
                                              const float* __restrict__ rspart,
                                              __bf16* __restrict__ agg) {
  size_t i = ((size_t)blockIdx.x * 256 + threadIdx.x) * 4;   // element index (B*N*D)
  size_t row = i >> 7;
  int d = (int)(i & (DD - 1));
  float inv = 1.0f / (rspart[row * 2 + 0] + rspart[row * 2 + 1] + 1e-6f);
  f32x4 p0 = *(const f32x4*)(aggpart + (row * 2 + 0) * DD + d);
  f32x4 p1 = *(const f32x4*)(aggpart + (row * 2 + 1) * DD + d);
  union { __bf16 h[4]; unsigned long long u; } o;
  o.h[0] = (__bf16)((p0.x + p1.x) * inv);
  o.h[1] = (__bf16)((p0.y + p1.y) * inv);
  o.h[2] = (__bf16)((p0.z + p1.z) * inv);
  o.h[3] = (__bf16)((p0.w + p1.w) * inv);
  *(unsigned long long*)(agg + i) = o.u;
}

// ---------------------------------------------------------------------------
// out = L2normalize(agg @ W_l + b_l + xg @ W_r), f32 output.
__global__ __launch_bounds__(256) void k_out(const __bf16* __restrict__ agg,
                                             const __bf16* __restrict__ xg,
                                             const __bf16* __restrict__ WlT,
                                             const __bf16* __restrict__ WrT,
                                             const float* __restrict__ bl,
                                             float* __restrict__ out) {
  __shared__ __bf16 sW[DD * DD];
  int wave = threadIdx.x >> 5, lane = threadIdx.x & 31;
  int ccol = lane & 15, crow = (lane >> 4) * 8;
  size_t m0 = (size_t)blockIdx.x * 128 + wave * 16;

  v16bf afa[4], afx[4];
#pragma unroll
  for (int kk = 0; kk < 4; kk++) {
    afa[kk] = load_afrag(agg + m0 * DD + kk * 32, DD);
    afx[kk] = load_afrag(xg + m0 * DD + kk * 32, DD);
  }
  v8f acc[8];

  for (int i = threadIdx.x; i < DD * DD / 8; i += 256)
    ((u32x4*)sW)[i] = ((const u32x4*)WlT)[i];
  __syncthreads();
#pragma unroll
  for (int dt = 0; dt < 8; dt++) {
    v8f c = {0.f, 0.f, 0.f, 0.f, 0.f, 0.f, 0.f, 0.f};
#pragma unroll
    for (int kk = 0; kk < 4; kk++)
      c = wmma_bf16(afa[kk], load_bfrag(sW + (dt * 16) * DD + kk * 32, DD), c);
    acc[dt] = c;
  }
  __syncthreads();
  for (int i = threadIdx.x; i < DD * DD / 8; i += 256)
    ((u32x4*)sW)[i] = ((const u32x4*)WrT)[i];
  __syncthreads();
  v8f ss = {0.f, 0.f, 0.f, 0.f, 0.f, 0.f, 0.f, 0.f};
#pragma unroll
  for (int dt = 0; dt < 8; dt++) {
    v8f c = acc[dt];
#pragma unroll
    for (int kk = 0; kk < 4; kk++)
      c = wmma_bf16(afx[kk], load_bfrag(sW + (dt * 16) * DD + kk * 32, DD), c);
    float bias = bl[dt * 16 + ccol];
#pragma unroll
    for (int r = 0; r < 8; r++) {
      float v = c[r] + bias;
      c[r] = v;
      ss[r] += v * v;
    }
    acc[dt] = c;
  }
#pragma unroll
  for (int r = 0; r < 8; r++) {
    float v = ss[r];
    v += __shfl_xor(v, 1, 32);
    v += __shfl_xor(v, 2, 32);
    v += __shfl_xor(v, 4, 32);
    v += __shfl_xor(v, 8, 32);
    ss[r] = 1.0f / fmaxf(__fsqrt_rn(v), 1e-12f);
  }
  float* op = out + m0 * OO;
#pragma unroll
  for (int dt = 0; dt < 8; dt++)
#pragma unroll
    for (int r = 0; r < 8; r++)
      op[(size_t)(crow + r) * OO + dt * 16 + ccol] = acc[dt][r] * ss[r];
}

// ---------------------------------------------------------------------------
extern "C" void kernel_launch(void* const* d_in, const int* in_sizes, int n_in,
                              void* d_out, int out_size, void* d_ws, size_t ws_size,
                              hipStream_t stream) {
  const float* x   = (const float*)d_in[0];
  const float* A   = (const float*)d_in[1];
  const float* Wqk = (const float*)d_in[2];
  const float* bqk = (const float*)d_in[3];
  const float* Wl  = (const float*)d_in[4];
  const float* bl  = (const float*)d_in[5];
  const float* Wr  = (const float*)d_in[6];
  const float* Wd  = (const float*)d_in[7];
  const float* bd  = (const float*)d_in[8];
  float* out = (float*)d_out;

  char* ws = (char*)d_ws;
  float*  deg  = (float*)ws;                                   ws += (size_t)BB * NN * 4;
  __bf16* WqkT = (__bf16*)ws;                                  ws += (size_t)DD * DD * 2;
  __bf16* WlT  = (__bf16*)ws;                                  ws += (size_t)DD * DD * 2;
  __bf16* WrT  = (__bf16*)ws;                                  ws += (size_t)DD * DD * 2;
  __bf16* xg   = (__bf16*)ws;                                  ws += (size_t)BB * NN * DD * 2;
  __bf16* xgT  = (__bf16*)ws;                                  ws += (size_t)BB * DD * NN * 2;
  __bf16* QK   = (__bf16*)ws;                                  ws += (size_t)BB * NN * DD * 2;
  __bf16* agg  = (__bf16*)ws;                                  ws += (size_t)BB * NN * DD * 2;
  float* aggpart = (float*)ws;                                 ws += (size_t)BB * NN * 2 * DD * 4;
  float* rspart  = (float*)ws;

  k_prep<<<3, 256, 0, stream>>>(Wqk, Wl, Wr, WqkT, WlT, WrT);
  k_deg<<<(BB * NN) / 8, 256, 0, stream>>>(A, deg);
  k_xg<<<(BB * NN * DD) / (4 * 256), 256, 0, stream>>>(x, deg, Wd, bd, xg, xgT);
  k_qk<<<(BB * NN) / 128, 256, 0, stream>>>(xg, WqkT, bqk, QK);
  k_attn<<<BB * 16 * 2, 256, 0, stream>>>(QK, xgT, A, aggpart, rspart);
  k_comb<<<(BB * NN * DD) / (4 * 256), 256, 0, stream>>>(aggpart, rspart, agg);
  k_out<<<(BB * NN) / 128, 256, 0, stream>>>(agg, xg, WlT, WrT, bl, out);
}
